// GraphSageRegression_86053964743052
// MI455X (gfx1250) — matrix operand order repaired and checked by reading, not compiled, not run on hardware
//
#include <hip/hip_runtime.h>
#include <stdint.h>

#define N_NODES 100000
#define N_EDGES 1600000
#define D 128

typedef __attribute__((ext_vector_type(16))) __bf16        v16bf;
typedef __attribute__((ext_vector_type(8)))  float         v8f;
typedef __attribute__((ext_vector_type(4)))  unsigned int  v4u;
typedef __attribute__((ext_vector_type(8)))  int           v8i;
typedef __attribute__((ext_vector_type(4)))  int           v4i;

// -------- float -> bf16 (RNE via hardware cast) --------
__device__ inline unsigned short f2bf(float f) {
    __bf16 b = (__bf16)f;
    return __builtin_bit_cast(unsigned short, b);
}

// -------- TDM: DMA a 16x128 f32 tile (row stride 128 floats) into LDS -------
// 6-arg clang-23 builtin: (g0 v4u, g1 v8i, g2 v4i, g3 v4i, extra v8i, cpol)
// Field packing per CDNA5 ISA.  2D tile: tile_dim0=128,
// tile_dim1=16, data_size=2 (4 bytes), tensor_dim0_stride=128, groups 2/3 zero.
__device__ inline void tdm_load_tile16x128(const float* gbase, unsigned ldsOff) {
    unsigned long long ga = (unsigned long long)(uintptr_t)gbase;
    v4u g0;
    g0.x = 1u;                                                 // count=1, user D#
    g0.y = ldsOff;                                             // lds_addr [63:32]
    g0.z = (unsigned)ga;                                       // global_addr[31:0]
    g0.w = (unsigned)((ga >> 32) & 0x01FFFFFFu) | (2u << 30);  // ga[56:32] | type=2
    v8i g1;
    g1[0] = (int)(2u << 16);    // workgroup_mask=0, data_size=2 (4B)
    g1[1] = (int)(128u << 16);  // tensor_dim0 = 128 (low16 at bits [63:48])
    g1[2] = (int)(16u << 16);   // tensor_dim0 hi=0; tensor_dim1 = 16
    g1[3] = (int)(128u << 16);  // tensor_dim1 hi=0; tile_dim0 = 128
    g1[4] = 16;                 // tile_dim1 = 16, tile_dim2 = 0
    g1[5] = 128;                // tensor_dim0_stride = 128 (low32)
    g1[6] = 0;                  // stride hi, tensor_dim1_stride lo16
    g1[7] = 0;                  // tensor_dim1_stride hi
    v4i z4 = {};
    v8i z8 = {};
    __builtin_amdgcn_tensor_load_to_lds(g0, g1, z4, z4, z8, 0);
}

// -------- weight fragment prep: swizzle 4 128x128 f32 matrices into
// WMMA B fragments (bf16, contiguous K, same K order as A): [mat(4)][nt(8)][kt(4)][lane(32)][j(16)]
// lane -> N column (lanes 0-15 / 16-31 = two K halves); k = kt*32+(lane>>4)*16+j
__global__ __launch_bounds__(256) void prep_frags(
    const float* __restrict__ Wl1, const float* __restrict__ Wr1,
    const float* __restrict__ Wl2, const float* __restrict__ Wr2,
    unsigned short* __restrict__ frag) {
    int tid  = blockIdx.x * 256 + threadIdx.x;   // 0..65535
    int mat  = tid >> 14;
    int rem  = tid & 16383;
    int nt   = rem >> 11;
    int rem2 = rem & 2047;
    int kt   = rem2 >> 9;
    int rem3 = rem2 & 511;
    int lane = rem3 >> 4;
    int j    = rem3 & 15;
    int k = kt * 32 + (lane >> 4) * 16 + j;
    int n = nt * 16 + (lane & 15);
    const float* W = (mat == 0) ? Wl1 : (mat == 1) ? Wr1 : (mat == 2) ? Wl2 : Wr2;
    frag[tid] = f2bf(W[k * D + n]);
}

// -------- degree count (once per launch; graph identical across layers) -----
__global__ __launch_bounds__(256) void degree_count(
    const long long* __restrict__ dst, float* __restrict__ deg) {
    int e = blockIdx.x * 256 + threadIdx.x;
    if (e < N_EDGES) atomicAdd(&deg[dst[e]], 1.0f);
}

// -------- edge scatter-add: one wave per edge, float4 per lane --------------
__global__ __launch_bounds__(256) void scatter_add(
    const long long* __restrict__ src, const long long* __restrict__ dst,
    const float* __restrict__ x, float* __restrict__ s) {
    const int t    = blockIdx.x * 256 + threadIdx.x;
    const int e    = t >> 5;              // uniform per wave -> scalar loads
    const int lane = t & 31;
    if (e >= N_EDGES) return;
    const long long sr = src[e];
    const long long ds = dst[e];
    const float4 v = ((const float4*)(x + (size_t)sr * D))[lane];
    float* o = s + (size_t)ds * D + lane * 4;
    atomicAdd(o + 0, v.x);
    atomicAdd(o + 1, v.y);
    atomicAdd(o + 2, v.z);
    atomicAdd(o + 3, v.w);
}

// -------- fused SAGE layer: relu( (s/deg)@Wl + x@Wr + b ) -------------------
// Block = 256 threads = 8 waves; block owns a 16-row M tile, wave w owns
// N-tile w.  TDM stages the two 16x128 f32 tiles into LDS; the 256 threads
// then build all bf16 A fragments once (tid -> (mat, kt, fragment-lane)),
// so each wave's K-loop is pure ds_load + global B-frag load + v_wmma.
__global__ __launch_bounds__(256) void sage_gemm(
    const float* __restrict__ s, const float* __restrict__ x,
    const float* __restrict__ deg,
    const unsigned short* __restrict__ fragL,
    const unsigned short* __restrict__ fragR,
    const float* __restrict__ bias, float* __restrict__ out, int relu) {
    __shared__ __align__(16) float tileS[16 * D];   // 8 KB
    __shared__ __align__(16) float tileX[16 * D];   // 8 KB
    __shared__ float ldsInv[16];
    __shared__ __align__(32) unsigned short afrag[2 * 4 * 32 * 16]; // 8 KB

    const int tid  = threadIdx.x;
    const int wave = tid >> 5;    // N-tile index 0..7
    const int lane = tid & 31;
    const int m0   = blockIdx.x * 16;               // 100000 = 6250*16, no tail

    if (wave == 0) {              // wave-uniform branch; TDM issued once/wave
        tdm_load_tile16x128(s + (size_t)m0 * D, (unsigned)(uintptr_t)tileS);
        tdm_load_tile16x128(x + (size_t)m0 * D, (unsigned)(uintptr_t)tileX);
        __builtin_amdgcn_s_wait_tensorcnt(0);
    }
    if (tid < 16) ldsInv[tid] = 1.0f / fmaxf(deg[m0 + tid], 1.0f);
    __syncthreads();

    // Stage A fragments (contiguous K layout): per fragment-lane fl
    // (row r = fl&15, half h = fl>>4): elems 0-15 = K kt*32+h*16+{0..15}.
    // 256 threads == 2 mats x 4 kt x 32 lanes exactly.
    {
        const int mat = tid >> 7;         // 0: aggregated (s/deg), 1: root (x)
        const int kt  = (tid >> 5) & 3;
        const int fl  = tid & 31;
        const int r   = fl & 15, h = fl >> 4;
        const int kb  = kt * 32 + h * 16;
        const float* rowp  = (mat == 0) ? (tileS + r * D) : (tileX + r * D);
        const float  scale = (mat == 0) ? ldsInv[r] : 1.0f;
        v16bf a;
#pragma unroll
        for (int j = 0; j < 8; ++j) {
            a[j]     = (__bf16)(rowp[kb + j] * scale);
            a[j + 8] = (__bf16)(rowp[kb + 8 + j] * scale);
        }
        *(v16bf*)(afrag + (size_t)tid * 16) = a;   // tid == ((mat*4+kt)*32+fl)
    }
    __syncthreads();

    v8f acc = {};
#pragma unroll
    for (int kt = 0; kt < 4; ++kt) {
        v16bf aL = *(const v16bf*)(afrag + (size_t)(kt * 32 + lane) * 16);
        v16bf aR = *(const v16bf*)(afrag + (size_t)((4 + kt) * 32 + lane) * 16);
        const size_t fo = (size_t)(((wave * 4 + kt) * 32 + lane) * 16);
        v16bf bL = *(const v16bf*)(fragL + fo);
        v16bf bR = *(const v16bf*)(fragR + fo);
        acc = __builtin_amdgcn_wmma_f32_16x16x32_bf16(
            false, aL, false, bL, (short)0, acc, false, false);
        acc = __builtin_amdgcn_wmma_f32_16x16x32_bf16(
            false, aR, false, bR, (short)0, acc, false, false);
    }

    // C/D layout: VGPR v -> M = v + h*8, N = lane&15
    const int r = lane & 15, h = lane >> 4;
    const int ncol = wave * 16 + r;
    const float bb = bias[ncol];
    float* obase = out + (size_t)m0 * D + ncol;
#pragma unroll
    for (int v = 0; v < 8; ++v) {
        float val = acc[v] + bb;
        if (relu) val = fmaxf(val, 0.0f);
        obase[(size_t)(v + h * 8) * D] = val;
    }
}

// -------- output layer (D_OUT=1): per-node dual dot + wave reduction --------
__global__ __launch_bounds__(256) void sage_out(
    const float* __restrict__ s, const float* __restrict__ hfeat,
    const float* __restrict__ deg,
    const float* __restrict__ Wl, const float* __restrict__ Wr,
    const float* __restrict__ b3, float* __restrict__ out) {
    const int t    = blockIdx.x * 256 + threadIdx.x;
    const int node = t >> 5;
    const int lane = t & 31;
    if (node >= N_NODES) return;
    const float inv = 1.0f / fmaxf(deg[node], 1.0f);
    const float4 a  = ((const float4*)(s + (size_t)node * D))[lane];
    const float4 c  = ((const float4*)(hfeat + (size_t)node * D))[lane];
    const float4 wl = ((const float4*)Wl)[lane];
    const float4 wr = ((const float4*)Wr)[lane];
    float acc = inv * (a.x * wl.x + a.y * wl.y + a.z * wl.z + a.w * wl.w)
              + c.x * wr.x + c.y * wr.y + c.z * wr.z + c.w * wr.w;
#pragma unroll
    for (int off = 16; off > 0; off >>= 1)
        acc += __shfl_xor(acc, off, 32);
    if (lane == 0) out[node] = acc + b3[0];
}

extern "C" void kernel_launch(void* const* d_in, const int* in_sizes, int n_in,
                              void* d_out, int out_size, void* d_ws, size_t ws_size,
                              hipStream_t stream) {
    (void)in_sizes; (void)n_in; (void)out_size; (void)ws_size;

    const float*     x   = (const float*)d_in[0];
    const long long* ei  = (const long long*)d_in[1];   // jnp.int64, [2, E]
    const float*     Wl1 = (const float*)d_in[2];
    const float*     Wr1 = (const float*)d_in[3];
    const float*     b1  = (const float*)d_in[4];
    const float*     Wl2 = (const float*)d_in[5];
    const float*     Wr2 = (const float*)d_in[6];
    const float*     b2  = (const float*)d_in[7];
    const float*     Wl3 = (const float*)d_in[8];
    const float*     Wr3 = (const float*)d_in[9];
    const float*     b3  = (const float*)d_in[10];
    const long long* src = ei;
    const long long* dst = ei + N_EDGES;

    // workspace layout
    char* ws = (char*)d_ws;
    const size_t featBytes = (size_t)N_NODES * D * sizeof(float);       // 51.2 MB
    float* s_buf = (float*)(ws);
    float* h1    = (float*)(ws + featBytes);
    float* h2    = (float*)(ws + 2 * featBytes);
    float* deg   = (float*)(ws + 3 * featBytes);
    size_t degBytes = (((size_t)N_NODES * sizeof(float)) + 255) & ~(size_t)255;
    unsigned short* frag = (unsigned short*)(ws + 3 * featBytes + degBytes);
    unsigned short* fWl1 = frag;
    unsigned short* fWr1 = frag + 16384;
    unsigned short* fWl2 = frag + 32768;
    unsigned short* fWr2 = frag + 49152;

    const int edgeWaveBlocks = (N_EDGES * 32) / 256;   // 200000
    const int edgeBlocks     = (N_EDGES + 255) / 256;  // 6250
    const int gemmBlocks     = N_NODES / 16;           // 6250 exact
    const int outBlocks      = (N_NODES * 32) / 256;   // 12500 exact

    // one-time (per launch) prep
    (void)hipMemsetAsync(deg, 0, (size_t)N_NODES * sizeof(float), stream);
    degree_count<<<edgeBlocks, 256, 0, stream>>>(dst, deg);
    prep_frags<<<256, 256, 0, stream>>>(Wl1, Wr1, Wl2, Wr2, frag);

    // layer 1: x -> h1
    (void)hipMemsetAsync(s_buf, 0, featBytes, stream);
    scatter_add<<<edgeWaveBlocks, 256, 0, stream>>>(src, dst, x, s_buf);
    sage_gemm<<<gemmBlocks, 256, 0, stream>>>(s_buf, x, deg, fWl1, fWr1, b1, h1, 1);

    // layer 2: h1 -> h2
    (void)hipMemsetAsync(s_buf, 0, featBytes, stream);
    scatter_add<<<edgeWaveBlocks, 256, 0, stream>>>(src, dst, h1, s_buf);
    sage_gemm<<<gemmBlocks, 256, 0, stream>>>(s_buf, h1, deg, fWl2, fWr2, b2, h2, 1);

    // layer 3: h2 -> out (D_OUT = 1)
    (void)hipMemsetAsync(s_buf, 0, featBytes, stream);
    scatter_add<<<edgeWaveBlocks, 256, 0, stream>>>(src, dst, h2, s_buf);
    sage_out<<<outBlocks, 256, 0, stream>>>(s_buf, h2, deg, Wl3, Wr3, b3, (float*)d_out);
}
